// minGPT_43207370998451
// MI455X (gfx1250) — compile-verified, matrix operand order checked by reading, not run on hardware
//
#include <hip/hip_runtime.h>

// ---------------------------------------------------------------------------
// GPT-2 forward (L=12, T=1024, C=1024, H=16, D=64) for gfx1250 (MI455X).
// GEMMs run on v_wmma_f32_16x16x32_f16 with a double-buffered LDS pipeline
// fed by GLOBAL_LOAD_ASYNC_TO_LDS (ASYNCcnt) when the toolchain exposes it.
// ---------------------------------------------------------------------------

typedef _Float16 v16h __attribute__((ext_vector_type(16)));
typedef float    v8f  __attribute__((ext_vector_type(8)));

#define TT 1024
#define CC 1024
#define HH 16
#define DH 64
#define LL 12
#define EPSF 1e-5f

#if defined(__has_builtin)
#if __has_builtin(__builtin_amdgcn_global_load_async_to_lds_b128) && \
    __has_builtin(__builtin_amdgcn_global_load_async_to_lds_b32) &&  \
    __has_builtin(__builtin_amdgcn_s_wait_asynccnt)
#define USE_ASYNC_LDS 1
#endif
#endif

#ifdef USE_ASYNC_LDS
// Exact parameter types per hipcc diagnostics:
//   b128: (v4i addrspace(1)*, v4i addrspace(3)*, imm int, imm int)
//   b32 : (int addrspace(1)*, int addrspace(3)*, imm int, imm int)
typedef int v4i_gcc __attribute__((vector_size(16)));
typedef __attribute__((address_space(1))) v4i_gcc* g128p;
typedef __attribute__((address_space(3))) v4i_gcc* l128p;
typedef __attribute__((address_space(1))) int*     g32p;
typedef __attribute__((address_space(3))) int*     l32p;
#define ASYNC_B128(g, l) \
  __builtin_amdgcn_global_load_async_to_lds_b128((g128p)(g), (l128p)(l), 0, 0)
#define ASYNC_B32(g, l) \
  __builtin_amdgcn_global_load_async_to_lds_b32((g32p)(g), (l32p)(l), 0, 0)
#endif

// ---------------------------------------------------------------------------
// Fragment builders. CDNA5 16-bit A/B operand layout: lane = hi*16 + r16
// holds K pairs at ((v>>2)<<4) + (hi<<3) + ((v&3)<<1).
// ---------------------------------------------------------------------------

// From f16 LDS tile [row][k] (paired 32-bit reads).
__device__ __forceinline__ v16h load_frag16(const _Float16* base, int stride,
                                            int row, int hi, int koff) {
  union { v16h v; unsigned int u[8]; } f;
#pragma unroll
  for (int r = 0; r < 8; ++r) {
    int kb = ((r >> 2) << 4) + (hi << 3) + ((r & 3) << 1) + koff;
    f.u[r] = *reinterpret_cast<const unsigned int*>(base + row * stride + kb);
  }
  return f.v;
}

// From fp32 LDS tile [row][k] (paired b64 reads + v_cvt_pk_f16_f32).
__device__ __forceinline__ v16h load_frag16_f32(const float* base, int stride,
                                                int row, int hi, int koff) {
  union { v16h v; _Float16 h[16]; } f;
#pragma unroll
  for (int r = 0; r < 8; ++r) {
    int kb = ((r >> 2) << 4) + (hi << 3) + ((r & 3) << 1) + koff;
    const float* p = base + row * stride + kb;
    f.h[2 * r]     = (_Float16)p[0];
    f.h[2 * r + 1] = (_Float16)p[1];
  }
  return f.v;
}

__device__ __forceinline__ float gelu_new_f(float x) {
  float x3 = x * x * x;
  return 0.5f * x * (1.0f + tanhf(0.7978845608028654f * (x + 0.044715f * x3)));
}

// ---------------------------------------------------------------------------
// LayerNorm: one block per row of C=1024 elements.
// ---------------------------------------------------------------------------
__global__ __launch_bounds__(256) void layernorm_kernel(
    const float* __restrict__ x, const float* __restrict__ g,
    const float* __restrict__ b, float* __restrict__ out) {
  __shared__ float rs[256];
  __shared__ float rs2[256];
  const int row = blockIdx.x;
  const int tid = threadIdx.x;
  float s = 0.0f, s2 = 0.0f;
  for (int i = tid; i < CC; i += 256) {
    float v = x[(size_t)row * CC + i];
    s += v; s2 += v * v;
  }
  rs[tid] = s; rs2[tid] = s2;
  __syncthreads();
  for (int off = 128; off > 0; off >>= 1) {
    if (tid < off) { rs[tid] += rs[tid + off]; rs2[tid] += rs2[tid + off]; }
    __syncthreads();
  }
  const float mu  = rs[0] * (1.0f / CC);
  const float var = rs2[0] * (1.0f / CC) - mu * mu;
  const float inv = rsqrtf(var + EPSF);
  for (int i = tid; i < CC; i += 256) {
    out[(size_t)row * CC + i] = (x[(size_t)row * CC + i] - mu) * inv * g[i] + b[i];
  }
}

// ---------------------------------------------------------------------------
// WMMA GEMM: Out[M,N] = epilogue(A[M,K] @ W[K,N] + bias[N]) (+ residual).
// 64x64 tile per workgroup; 8 waves, each owning two 16x16 WMMA tiles.
// fp32 tiles staged in double-buffered LDS via async global->LDS copies
// (per-lane addressing also performs the K-transpose of W during the copy);
// fp32 -> f16 conversion happens in the fragment build and co-executes with
// the WMMAs. fuse_gelu: 1 -> tanh-approx GELU epilogue.
// ---------------------------------------------------------------------------
__global__ __launch_bounds__(256) void wmma_gemm_kernel(
    const float* __restrict__ A, const float* __restrict__ W,
    const float* __restrict__ bias, const float* __restrict__ resid,
    float* __restrict__ Out, int M, int N, int K, int fuse_gelu) {
  // stride 36 floats = 144 B rows: keeps 16 B alignment for b128 async stores
  // and 8 B alignment for the paired fragment reads.
  __shared__ float Ash[2][64][36];   // [buf][m][k]
  __shared__ float BshT[2][64][36];  // [buf][n][k]  (W tile K-transposed)

  const int tid  = threadIdx.x;
  const int lane = tid & 31;
  const int w    = tid >> 5;
  const int hi   = lane >> 4;
  const int r16  = lane & 15;
  const int m0   = blockIdx.y * 64;
  const int n0   = blockIdx.x * 64;
  const int rt0  = (w >> 2) * 16;   // wave's first row-tile (0 or 16)
  const int rt1  = rt0 + 32;        // wave's second row-tile (32 or 48)
  const int nb   = (w & 3) * 16;    // wave's col-tile

  v8f acc0 = {};
  v8f acc1 = {};

  const int arow = tid >> 2, acb = (tid & 3) * 8;  // A: 4 thr/row x 8 floats
  const int bk   = tid >> 3, bnb = (tid & 7) * 8;  // W: 8 thr/row x 8 floats

  auto issue_tile = [&](int buf, int kt) {
#ifdef USE_ASYNC_LDS
    const float* gA = &A[(size_t)(m0 + arow) * K + kt + acb];
    float* lA = &Ash[buf][arow][acb];
    ASYNC_B128(gA, lA);           // 2 x 16B per lane covers the 64x32 A tile
    ASYNC_B128(gA + 4, lA + 4);
#pragma unroll
    for (int i = 0; i < 8; ++i)   // per-lane scatter = transposed W copy
      ASYNC_B32(&W[(size_t)(kt + bk) * N + n0 + bnb + i],
                &BshT[buf][bnb + i][bk]);
#else
#pragma unroll
    for (int i = 0; i < 8; ++i)
      Ash[buf][arow][acb + i] = A[(size_t)(m0 + arow) * K + kt + acb + i];
#pragma unroll
    for (int i = 0; i < 8; ++i)
      BshT[buf][bnb + i][bk] = W[(size_t)(kt + bk) * N + n0 + bnb + i];
    if (kt + 32 < K)
      __builtin_prefetch(&W[(size_t)(kt + 32 + bk) * N + n0 + bnb], 0, 1);
#endif
  };

  const int nchunks = K >> 5;
  issue_tile(0, 0);

  for (int c = 0; c < nchunks; ++c) {
    __syncthreads();  // previous compute done reading buf (c+1)&1
    if (c + 1 < nchunks) {
      issue_tile((c + 1) & 1, (c + 1) << 5);
#ifdef USE_ASYNC_LDS
      __builtin_amdgcn_s_wait_asynccnt(10);  // current buffer's 10 copies done
#endif
    } else {
#ifdef USE_ASYNC_LDS
      __builtin_amdgcn_s_wait_asynccnt(0);
#endif
    }
    __syncthreads();  // all waves' copies for buf c&1 visible

    const int buf = c & 1;
    v16h bf = load_frag16_f32(&BshT[buf][0][0], 36, nb + r16, hi, 0);
    v16h a0 = load_frag16_f32(&Ash[buf][0][0],  36, rt0 + r16, hi, 0);
    v16h a1 = load_frag16_f32(&Ash[buf][0][0],  36, rt1 + r16, hi, 0);
    acc0 = __builtin_amdgcn_wmma_f32_16x16x32_f16(false, a0, false, bf,
                                                  (short)0, acc0, false, false);
    acc1 = __builtin_amdgcn_wmma_f32_16x16x32_f16(false, a1, false, bf,
                                                  (short)0, acc1, false, false);
  }

  const int gn = n0 + nb + r16;
  const float bn = bias[gn];
#pragma unroll
  for (int r = 0; r < 8; ++r) {
    {
      int gm = m0 + rt0 + r + hi * 8;
      float v = acc0[r] + bn;
      if (fuse_gelu) v = gelu_new_f(v);
      if (resid) v += resid[(size_t)gm * N + gn];
      Out[(size_t)gm * N + gn] = v;
    }
    {
      int gm = m0 + rt1 + r + hi * 8;
      float v = acc1[r] + bn;
      if (fuse_gelu) v = gelu_new_f(v);
      if (resid) v += resid[(size_t)gm * N + gn];
      Out[(size_t)gm * N + gn] = v;
    }
  }
}

// ---------------------------------------------------------------------------
// Flash attention: one workgroup per (query-block of 64, head).
// S = (Q/8) K^T via WMMA, online softmax in fp32 LDS, O += P V via WMMA.
// qkv layout: [T][3C] with q at col h*64, k at C+h*64, v at 2C+h*64.
// ---------------------------------------------------------------------------
__global__ __launch_bounds__(256) void flash_attn_kernel(
    const float* __restrict__ qkv, const float* __restrict__ amask,
    float* __restrict__ y) {
  __shared__ _Float16 Qs [64][68];  // [m][d]   (pre-scaled by 1/sqrt(D))
  __shared__ _Float16 Ks [64][68];  // [key][d] (== B^T layout for QK^T)
  __shared__ _Float16 Ps [64][68];  // [m][key] f16 probabilities
  __shared__ _Float16 VsT[64][68];  // [d][key] (B^T layout for P@V)
  __shared__ float    Ss [64][65];  // fp32 scores block
  __shared__ float    rowm[64], rowl[64], alf[64];

  const int tid  = threadIdx.x;
  const int lane = tid & 31;
  const int w    = tid >> 5;
  const int hi   = lane >> 4;
  const int r16  = lane & 15;
  const int q0   = blockIdx.x * 64;
  const int h    = blockIdx.y;
  const int qoff = h * DH;
  const int koff = CC + h * DH;
  const int voff = 2 * CC + h * DH;
  const int rt0  = (w >> 2) * 16, rt1 = rt0 + 32, nb = (w & 3) * 16;

  {  // load Q block once, pre-scaled
    int row = tid >> 2, cb = (tid & 3) * 16;
#pragma unroll
    for (int i = 0; i < 16; ++i)
      Qs[row][cb + i] =
          (_Float16)(qkv[(size_t)(q0 + row) * (3 * CC) + qoff + cb + i] * 0.125f);
  }
  if (tid < 64) { rowm[tid] = -3.0e38f; rowl[tid] = 0.0f; }

  v8f o0 = {};
  v8f o1 = {};

  for (int kb = 0; kb < TT / 64; ++kb) {
    const int k0 = kb * 64;
    __syncthreads();  // protect Ks/VsT/Ss/Ps reuse from previous iteration
    {
      int row = tid >> 2, cb = (tid & 3) * 16;
#pragma unroll
      for (int i = 0; i < 16; ++i)
        Ks[row][cb + i] =
            (_Float16)qkv[(size_t)(k0 + row) * (3 * CC) + koff + cb + i];
#pragma unroll
      for (int i = 0; i < 16; ++i)
        VsT[cb + i][row] =
            (_Float16)qkv[(size_t)(k0 + row) * (3 * CC) + voff + cb + i];
    }
    __syncthreads();

    // S tile = Q K^T over d=64 (two K-chunks of 32)
    v8f s0 = {};
    v8f s1 = {};
#pragma unroll
    for (int kk = 0; kk < 64; kk += 32) {
      v16h bf = load_frag16(&Ks[0][0], 68, nb + r16, hi, kk);
      v16h a0 = load_frag16(&Qs[0][0], 68, rt0 + r16, hi, kk);
      v16h a1 = load_frag16(&Qs[0][0], 68, rt1 + r16, hi, kk);
      s0 = __builtin_amdgcn_wmma_f32_16x16x32_f16(false, a0, false, bf,
                                                  (short)0, s0, false, false);
      s1 = __builtin_amdgcn_wmma_f32_16x16x32_f16(false, a1, false, bf,
                                                  (short)0, s1, false, false);
    }
    const float mb = -1e9f * (1.0f - amask[k0 + nb + r16]);
#pragma unroll
    for (int r = 0; r < 8; ++r) {
      Ss[rt0 + r + hi * 8][nb + r16] = s0[r] + mb;
      Ss[rt1 + r + hi * 8][nb + r16] = s1[r] + mb;
    }
    __syncthreads();

    // online softmax update (one thread per query row)
    if (tid < 64) {
      float mold = rowm[tid];
      float mx = mold;
      for (int j = 0; j < 64; ++j) mx = fmaxf(mx, Ss[tid][j]);
      float a = __expf(mold - mx);
      float lsum = 0.0f;
      for (int j = 0; j < 64; ++j) {
        float p = __expf(Ss[tid][j] - mx);
        Ps[tid][j] = (_Float16)p;
        lsum += p;
      }
      rowm[tid] = mx;
      rowl[tid] = rowl[tid] * a + lsum;
      alf[tid] = a;
    }
    __syncthreads();

    // rescale accumulators, then O += P V
#pragma unroll
    for (int r = 0; r < 8; ++r) {
      o0[r] *= alf[rt0 + r + hi * 8];
      o1[r] *= alf[rt1 + r + hi * 8];
    }
#pragma unroll
    for (int kk = 0; kk < 64; kk += 32) {
      v16h bf = load_frag16(&VsT[0][0], 68, nb + r16, hi, kk);
      v16h a0 = load_frag16(&Ps[0][0],  68, rt0 + r16, hi, kk);
      v16h a1 = load_frag16(&Ps[0][0],  68, rt1 + r16, hi, kk);
      o0 = __builtin_amdgcn_wmma_f32_16x16x32_f16(false, a0, false, bf,
                                                  (short)0, o0, false, false);
      o1 = __builtin_amdgcn_wmma_f32_16x16x32_f16(false, a1, false, bf,
                                                  (short)0, o1, false, false);
    }
  }
  __syncthreads();

#pragma unroll
  for (int r = 0; r < 8; ++r) {
    int m = rt0 + r + hi * 8;
    y[(size_t)(q0 + m) * CC + h * DH + nb + r16] = o0[r] / rowl[m];
    int m2 = rt1 + r + hi * 8;
    y[(size_t)(q0 + m2) * CC + h * DH + nb + r16] = o1[r] / rowl[m2];
  }
}

// ---------------------------------------------------------------------------
// Driver
// ---------------------------------------------------------------------------
extern "C" void kernel_launch(void* const* d_in, const int* in_sizes, int n_in,
                              void* d_out, int out_size, void* d_ws, size_t ws_size,
                              hipStream_t stream) {
  (void)in_sizes; (void)n_in; (void)out_size; (void)ws_size;
  const float* embeds = (const float*)d_in[0];
  const float* amask  = (const float*)d_in[1];
  const float* ln1g   = (const float*)d_in[2];
  const float* ln1b   = (const float*)d_in[3];
  const float* Wqkv   = (const float*)d_in[4];
  const float* bqkv   = (const float*)d_in[5];
  const float* Wo     = (const float*)d_in[6];
  const float* bo     = (const float*)d_in[7];
  const float* ln2g   = (const float*)d_in[8];
  const float* ln2b   = (const float*)d_in[9];
  const float* Wfc    = (const float*)d_in[10];
  const float* bfc    = (const float*)d_in[11];
  const float* Wmp    = (const float*)d_in[12];
  const float* bmp    = (const float*)d_in[13];
  const float* lnfg   = (const float*)d_in[14];
  const float* lnfb   = (const float*)d_in[15];

  float* ws = (float*)d_ws;
  const size_t TC = (size_t)TT * CC;
  float* xb  = ws;             // residual stream  [T,C]
  float* hb  = xb + TC;        // LN output        [T,C]
  float* qkb = hb + TC;        // qkv              [T,3C]
  float* yb  = qkb + 3 * TC;   // attention output [T,C]
  float* fb  = yb + TC;        // MLP hidden       [T,4C]

  (void)hipMemcpyAsync(xb, embeds, TC * sizeof(float),
                       hipMemcpyDeviceToDevice, stream);

  dim3 blk(256);
  for (int l = 0; l < LL; ++l) {
    layernorm_kernel<<<TT, blk, 0, stream>>>(
        xb, ln1g + (size_t)l * CC, ln1b + (size_t)l * CC, hb);
    wmma_gemm_kernel<<<dim3(3 * CC / 64, TT / 64), blk, 0, stream>>>(
        hb, Wqkv + (size_t)l * CC * 3 * CC, bqkv + (size_t)l * 3 * CC,
        nullptr, qkb, TT, 3 * CC, CC, 0);
    flash_attn_kernel<<<dim3(TT / 64, HH), blk, 0, stream>>>(qkb, amask, yb);
    wmma_gemm_kernel<<<dim3(CC / 64, TT / 64), blk, 0, stream>>>(
        yb, Wo + (size_t)l * CC * CC, bo + (size_t)l * CC,
        xb, xb, TT, CC, CC, 0);
    layernorm_kernel<<<TT, blk, 0, stream>>>(
        xb, ln2g + (size_t)l * CC, ln2b + (size_t)l * CC, hb);
    wmma_gemm_kernel<<<dim3(4 * CC / 64, TT / 64), blk, 0, stream>>>(
        hb, Wfc + (size_t)l * CC * 4 * CC, bfc + (size_t)l * 4 * CC,
        nullptr, fb, TT, 4 * CC, CC, 1);
    wmma_gemm_kernel<<<dim3(CC / 64, TT / 64), blk, 0, stream>>>(
        fb, Wmp + (size_t)l * 4 * CC * CC, bmp + (size_t)l * CC,
        xb, xb, TT, CC, 4 * CC, 0);
  }
  layernorm_kernel<<<TT, blk, 0, stream>>>(xb, lnfg, lnfb, (float*)d_out);
}
